// MPNetLayer_46566035423309
// MI455X (gfx1250) — matrix operand order repaired and checked by reading, not compiled, not run on hardware
//
#include <hip/hip_runtime.h>
#include <hip/hip_bf16.h>

// ---------------------------------------------------------------------------
// MPNet encoder layer, f16-WMMA implementation for gfx1250 (CDNA5, wave32).
// B=2, S=2048, E=1024, H=16, DH=64, FF=4096.
// ---------------------------------------------------------------------------

typedef _Float16 v16h __attribute__((ext_vector_type(16)));
typedef float    v8f  __attribute__((ext_vector_type(8)));

#define E_DIM 1024
#define H_DIM 16
#define DH_DIM 64
#define FF_DIM 4096
#define B_DIM 2
#define S_DIM 2048
#define M_DIM (B_DIM * S_DIM)   // 4096 rows

union FragA { v16h v; unsigned u[8]; };
union PairCv { unsigned u; _Float16 h[2]; };

// A-matrix 16x32 f16 fragment K index for element pair p (e = 2p), per ISA 7.12.2:
//   p in [0,4): K = 2p + 8*half ; p in [4,8): K = 2p + 8 + 8*half
__device__ __forceinline__ int a_pair_k(int p, int half) {
    return 2 * p + 8 * half + ((p >= 4) ? 8 : 0);
}
// B-matrix 32x16 f16 fragment K index for element pair p (same order as A):
//   p in [0,4): K = 2p + 8*half ; p in [4,8): K = 2p + 8 + 8*half
__device__ __forceinline__ int b_pair_k(int p, int half) {
    return 2 * p + 8 * half + ((p >= 4) ? 8 : 0);
}

// CDNA5 async global->LDS dword copy (ASYNCcnt tracked). lds_addr is the
// 32-bit LDS byte address (low 32 bits of the flat shared-aperture address).
__device__ __forceinline__ void async_ld_dword(void* lds_ptr, const void* gptr) {
    unsigned lds_addr = (unsigned)(size_t)lds_ptr;
    unsigned long long ga = (unsigned long long)(size_t)gptr;
    asm volatile("global_load_async_to_lds_b32 %0, %1, off"
                 :: "v"(lds_addr), "v"(ga)
                 : "memory");
}
__device__ __forceinline__ void wait_async_all() {
    asm volatile("s_wait_asynccnt 0x0" ::: "memory");
}

__device__ __forceinline__ float rmax16(float v) {
    v = fmaxf(v, __shfl_xor(v, 1, 32));
    v = fmaxf(v, __shfl_xor(v, 2, 32));
    v = fmaxf(v, __shfl_xor(v, 4, 32));
    v = fmaxf(v, __shfl_xor(v, 8, 32));
    return v;
}
__device__ __forceinline__ float rsum16(float v) {
    v += __shfl_xor(v, 1, 32);
    v += __shfl_xor(v, 2, 32);
    v += __shfl_xor(v, 4, 32);
    v += __shfl_xor(v, 8, 32);
    return v;
}

__device__ __forceinline__ float gelu_exact(float x) {
    return 0.5f * x * (1.0f + erff(x * 0.70710678118654752f));
}

// ---------------------------------------------------------------------------
// fp32 -> f16 cast
// ---------------------------------------------------------------------------
__global__ void cast_f16_kernel(const float* __restrict__ src,
                                _Float16* __restrict__ dst, int n) {
    int i = blockIdx.x * blockDim.x + threadIdx.x;
    if (i < n) dst[i] = (_Float16)src[i];
}

// ---------------------------------------------------------------------------
// Tiled WMMA GEMM:  C[M,N] = act(A[M,K] @ W[K,N] + bias)
// A, W f16 row-major; fp32 accumulate; optional fp32 and f16 outputs.
// Block = 256 threads (8 waves), 128x64 C tile, K staged in 32-chunks via LDS.
// A tile staged with CDNA5 async-to-LDS; B tile staged transposed (regular).
// ---------------------------------------------------------------------------
__global__ void gemm_f16_kernel(const _Float16* __restrict__ A,
                                const _Float16* __restrict__ W,
                                const float* __restrict__ bias,
                                float* __restrict__ Cf,
                                _Float16* __restrict__ Ch,
                                int Mdim, int Ndim, int Kdim, int act) {
    __shared__ alignas(16) _Float16 As[128 * 32];  // [row][k]
    __shared__ alignas(16) _Float16 Bs[64 * 32];   // transposed: [n][k]

    const int tid  = threadIdx.x;
    const int wave = tid >> 5;
    const int lane = tid & 31;
    const int ln   = lane & 15;
    const int half = lane >> 4;

    const int n0 = blockIdx.x * 64;
    const int m0 = blockIdx.y * 128;

    v8f acc[4];
    for (int t = 0; t < 4; ++t)
        for (int r = 0; r < 8; ++r) acc[t][r] = 0.0f;

    for (int kc = 0; kc < Kdim; kc += 32) {
        // stage A tile (128 rows x 32 k) via async dword copies
        for (int i = tid; i < 2048; i += 256) {
            int r  = i >> 4;
            int kk = (i & 15) << 1;
            async_ld_dword(&As[r * 32 + kk],
                           &A[(size_t)(m0 + r) * Kdim + kc + kk]);
        }
        // stage W tile (32 k x 64 n) transposed into Bs[n][k]
        for (int i = tid; i < 1024; i += 256) {
            int kk = i >> 5;
            int nn = (i & 31) << 1;
            PairCv cv;
            cv.u = *(const unsigned*)&W[(size_t)(kc + kk) * Ndim + n0 + nn];
            Bs[nn * 32 + kk]       = cv.h[0];
            Bs[(nn + 1) * 32 + kk] = cv.h[1];
        }
        wait_async_all();
        __syncthreads();

        FragA af;
        const int arow = wave * 16 + ln;
        for (int p = 0; p < 8; ++p)
            af.u[p] = *(const unsigned*)&As[arow * 32 + a_pair_k(p, half)];

        for (int t = 0; t < 4; ++t) {
            FragA bf;
            const int brow = t * 16 + ln;
            for (int p = 0; p < 8; ++p)
                bf.u[p] = *(const unsigned*)&Bs[brow * 32 + b_pair_k(p, half)];
            acc[t] = __builtin_amdgcn_wmma_f32_16x16x32_f16(
                false, af.v, false, bf.v, (short)0, acc[t], false, false);
        }
        __syncthreads();
    }

    // epilogue: bias (+gelu), dual-precision store
    for (int t = 0; t < 4; ++t) {
        const int gn = n0 + t * 16 + ln;
        const float bv = bias[gn];
        for (int r = 0; r < 8; ++r) {
            const int gm = m0 + wave * 16 + r + 8 * half;
            float vv = acc[t][r] + bv;
            if (act) vv = gelu_exact(vv);
            const size_t idx = (size_t)gm * Ndim + gn;
            if (Cf) Cf[idx] = vv;
            if (Ch) Ch[idx] = (_Float16)vv;
        }
    }
}

// ---------------------------------------------------------------------------
// Flash attention. Block = 128 threads (4 waves), one (b,h) per block; each
// wave owns one 16-query tile. Loop over 32-key tiles; K/V tiles staged
// cooperatively in LDS (shared by all 4 waves): K via async-to-LDS,
// V transposed with regular loads. scores = (Q K^T)/8 + pb + mask;
// online softmax; ctx = P V with P staged per-wave through LDS for the
// C-layout -> A-layout fragment conversion.
// ---------------------------------------------------------------------------
__global__ void attn_kernel(const _Float16* __restrict__ q,
                            const _Float16* __restrict__ k,
                            const _Float16* __restrict__ v,
                            const float* __restrict__ pb,
                            const float* __restrict__ mask,
                            _Float16* __restrict__ ctx) {
    __shared__ alignas(16) _Float16 Ks[32 * 64];     // [key][dh]
    __shared__ alignas(16) _Float16 Vt[64 * 32];     // [dh][key]
    __shared__ alignas(16) _Float16 Ps[4][16 * 32];  // per-wave P tile

    const int tid  = threadIdx.x;
    const int wave = tid >> 5;
    const int lane = tid & 31;
    const int ln   = lane & 15;
    const int half = lane >> 4;

    const int b = blockIdx.z;
    const int h = blockIdx.y;
    const int qbase = (blockIdx.x * 4 + wave) * 16;

    // load Q fragments for dh chunks [0,32) and [32,64)
    FragA aq[2];
    {
        const size_t qrowbase = ((size_t)(b * S_DIM + qbase + ln)) * E_DIM + h * DH_DIM;
        for (int c = 0; c < 2; ++c)
            for (int p = 0; p < 8; ++p)
                aq[c].u[p] = *(const unsigned*)&q[qrowbase + a_pair_k(p, half) + 32 * c];
    }

    float mrow[8], lrow[8];
    v8f acc[4];
    for (int r = 0; r < 8; ++r) { mrow[r] = -1e30f; lrow[r] = 0.0f; }
    for (int t = 0; t < 4; ++t)
        for (int r = 0; r < 8; ++r) acc[t][r] = 0.0f;

    for (int kt = 0; kt < S_DIM; kt += 32) {
        // ---- cooperative stage of K tile [32 keys x 64 dh] (async) ----
        for (int i = tid; i < 1024; i += 128) {
            int key = i >> 5;
            int dh  = (i & 31) << 1;
            async_ld_dword(&Ks[key * 64 + dh],
                           &k[((size_t)(b * S_DIM + kt + key)) * E_DIM + h * DH_DIM + dh]);
        }
        // ---- cooperative stage of V tile transposed -> Vt[dh][key] ----
        for (int i = tid; i < 1024; i += 128) {
            int key = i >> 5;
            int dh  = (i & 31) << 1;
            PairCv cv;
            cv.u = *(const unsigned*)&v[((size_t)(b * S_DIM + kt + key)) * E_DIM +
                                        h * DH_DIM + dh];
            Vt[dh * 32 + key]       = cv.h[0];
            Vt[(dh + 1) * 32 + key] = cv.h[1];
        }
        // prefetch next iteration's position-bias tile
        if (kt + 32 < S_DIM) {
            const float* nb = &pb[((size_t)h * S_DIM + qbase + ln) * S_DIM +
                                  kt + 32 + half * 16];
            __builtin_prefetch(nb, 0, 1);
        }
        wait_async_all();
        __syncthreads();

        // ---- scores: 16 x 32 tile as two 16x16 fragments ----
        v8f s[2];
        for (int sub = 0; sub < 2; ++sub) {
            for (int r = 0; r < 8; ++r) s[sub][r] = 0.0f;
            const int keyl = sub * 16 + ln;
            for (int c = 0; c < 2; ++c) {
                FragA bk;
                for (int p = 0; p < 8; ++p)
                    bk.u[p] = *(const unsigned*)&Ks[keyl * 64 + b_pair_k(p, half) + 32 * c];
                s[sub] = __builtin_amdgcn_wmma_f32_16x16x32_f16(
                    false, aq[c].v, false, bk.v, (short)0, s[sub], false, false);
            }
        }
        // ---- scale + position bias + mask ----
        for (int sub = 0; sub < 2; ++sub) {
            const int key = kt + sub * 16 + ln;
            const float am = mask[b * S_DIM + key];
            for (int r = 0; r < 8; ++r) {
                const int qrow = qbase + r + 8 * half;
                s[sub][r] = s[sub][r] * 0.125f +
                            pb[((size_t)h * S_DIM + qrow) * S_DIM + key] + am;
            }
        }
        // ---- online softmax (row stats across the 16-lane group) ----
        for (int r = 0; r < 8; ++r) {
            float mx = rmax16(fmaxf(s[0][r], s[1][r]));
            float mn = fmaxf(mrow[r], mx);
            float scl = expf(mrow[r] - mn);
            float p0 = expf(s[0][r] - mn);
            float p1 = expf(s[1][r] - mn);
            lrow[r] = lrow[r] * scl + rsum16(p0 + p1);
            mrow[r] = mn;
            for (int t = 0; t < 4; ++t) acc[t][r] *= scl;
            const int prow = r + 8 * half;
            Ps[wave][prow * 32 + ln]      = (_Float16)p0;
            Ps[wave][prow * 32 + 16 + ln] = (_Float16)p1;
        }
        // wait for per-wave LDS writes (cross-lane layout conversion)
        asm volatile("s_wait_dscnt 0x0" ::: "memory");

        // ---- re-read P in A-fragment layout ----
        FragA ap;
        for (int p = 0; p < 8; ++p)
            ap.u[p] = *(const unsigned*)&Ps[wave][ln * 32 + a_pair_k(p, half)];
        asm volatile("" ::: "memory");

        // ---- ctx += P[16x32] @ V[32x64] ----
        for (int t = 0; t < 4; ++t) {
            FragA bv;
            const int dhl = t * 16 + ln;
            for (int p = 0; p < 8; ++p)
                bv.u[p] = *(const unsigned*)&Vt[dhl * 32 + b_pair_k(p, half)];
            acc[t] = __builtin_amdgcn_wmma_f32_16x16x32_f16(
                false, ap.v, false, bv.v, (short)0, acc[t], false, false);
        }
        __syncthreads();   // protect Ks/Vt before next restage
    }

    // ---- normalize + store ctx (f16) ----
    for (int t = 0; t < 4; ++t)
        for (int r = 0; r < 8; ++r) {
            const float ov = acc[t][r] / lrow[r];
            const size_t idx = ((size_t)(b * S_DIM + qbase + r + 8 * half)) * E_DIM +
                               h * DH_DIM + t * 16 + ln;
            ctx[idx] = (_Float16)ov;
        }
}

// ---------------------------------------------------------------------------
// out = LayerNorm(x + y) * g + b ; optional fp32 and f16 outputs.
// One block (256 threads) per row of E=1024.
// ---------------------------------------------------------------------------
__global__ void add_ln_kernel(const float* __restrict__ x,
                              const float* __restrict__ y,
                              const float* __restrict__ g,
                              const float* __restrict__ bta,
                              float* __restrict__ of,
                              _Float16* __restrict__ oh) {
    __shared__ float red[8];
    __shared__ float s_mu, s_rstd;

    const int tid = threadIdx.x;
    const int wave = tid >> 5;
    const int lane = tid & 31;
    const size_t base = (size_t)blockIdx.x * E_DIM;

    float vals[4];
    float s = 0.0f;
    for (int i = 0; i < 4; ++i) {
        const int col = tid + i * 256;
        const float t = x[base + col] + y[base + col];
        vals[i] = t;
        s += t;
    }
    for (int off = 16; off > 0; off >>= 1) s += __shfl_down(s, off, 32);
    if (lane == 0) red[wave] = s;
    __syncthreads();
    if (tid == 0) {
        float tot = 0.0f;
        for (int j = 0; j < 8; ++j) tot += red[j];
        s_mu = tot * (1.0f / E_DIM);
    }
    __syncthreads();
    const float mu = s_mu;

    float vs = 0.0f;
    for (int i = 0; i < 4; ++i) {
        const float d = vals[i] - mu;
        vs += d * d;
    }
    for (int off = 16; off > 0; off >>= 1) vs += __shfl_down(vs, off, 32);
    __syncthreads();
    if (lane == 0) red[wave] = vs;
    __syncthreads();
    if (tid == 0) {
        float tot = 0.0f;
        for (int j = 0; j < 8; ++j) tot += red[j];
        s_rstd = rsqrtf(tot * (1.0f / E_DIM) + 1e-5f);
    }
    __syncthreads();
    const float rstd = s_rstd;

    for (int i = 0; i < 4; ++i) {
        const int col = tid + i * 256;
        const float o = (vals[i] - mu) * rstd * g[col] + bta[col];
        if (of) of[base + col] = o;
        if (oh) oh[base + col] = (_Float16)o;
    }
}

// ---------------------------------------------------------------------------
// launch
// ---------------------------------------------------------------------------
extern "C" void kernel_launch(void* const* d_in, const int* in_sizes, int n_in,
                              void* d_out, int out_size, void* d_ws, size_t ws_size,
                              hipStream_t stream) {
    (void)in_sizes; (void)n_in; (void)out_size; (void)ws_size;

    const float* x    = (const float*)d_in[0];
    const float* amask= (const float*)d_in[1];
    const float* pb   = (const float*)d_in[2];
    const float* Wq   = (const float*)d_in[3];
    const float* bq   = (const float*)d_in[4];
    const float* Wk   = (const float*)d_in[5];
    const float* bk   = (const float*)d_in[6];
    const float* Wv   = (const float*)d_in[7];
    const float* bv   = (const float*)d_in[8];
    const float* Wo   = (const float*)d_in[9];
    const float* bo   = (const float*)d_in[10];
    const float* ln1g = (const float*)d_in[11];
    const float* ln1b = (const float*)d_in[12];
    const float* W1   = (const float*)d_in[13];
    const float* b1   = (const float*)d_in[14];
    const float* W2   = (const float*)d_in[15];
    const float* b2   = (const float*)d_in[16];
    const float* ln2g = (const float*)d_in[17];
    const float* ln2b = (const float*)d_in[18];

    char* w = (char*)d_ws;
    size_t off = 0;
    auto take = [&](size_t bytes) -> void* {
        void* p = w + off;
        off += (bytes + 255) & ~(size_t)255;
        return p;
    };
    const size_t ME  = (size_t)M_DIM * E_DIM;
    const size_t MF  = (size_t)M_DIM * FF_DIM;
    const size_t EE  = (size_t)E_DIM * E_DIM;
    const size_t EF  = (size_t)E_DIM * FF_DIM;

    _Float16* xh   = (_Float16*)take(ME * 2);
    _Float16* wqh  = (_Float16*)take(EE * 2);
    _Float16* wkh  = (_Float16*)take(EE * 2);
    _Float16* wvh  = (_Float16*)take(EE * 2);
    _Float16* woh  = (_Float16*)take(EE * 2);
    _Float16* w1h  = (_Float16*)take(EF * 2);
    _Float16* w2h  = (_Float16*)take(EF * 2);
    _Float16* qh   = (_Float16*)take(ME * 2);
    _Float16* kh   = (_Float16*)take(ME * 2);
    _Float16* vh   = (_Float16*)take(ME * 2);
    _Float16* ctxh = (_Float16*)take(ME * 2);
    float*    attnf= (float*)take(ME * 4);
    float*    hf   = (float*)take(ME * 4);
    _Float16* hh   = (_Float16*)take(ME * 2);
    _Float16* f1h  = (_Float16*)take(MF * 2);
    float*    f2f  = (float*)take(ME * 4);

    auto cast = [&](const float* s, _Float16* d, size_t n) {
        cast_f16_kernel<<<dim3((unsigned)((n + 255) / 256)), dim3(256), 0, stream>>>(
            s, d, (int)n);
    };
    cast(x,  xh,  ME);
    cast(Wq, wqh, EE);
    cast(Wk, wkh, EE);
    cast(Wv, wvh, EE);
    cast(Wo, woh, EE);
    cast(W1, w1h, EF);
    cast(W2, w2h, EF);

    const dim3 gemm_blk(256);
    const dim3 g_e(E_DIM / 64, M_DIM / 128);   // N=1024 tiles
    const dim3 g_f(FF_DIM / 64, M_DIM / 128);  // N=4096 tiles

    // QKV projections (f16 out only)
    gemm_f16_kernel<<<g_e, gemm_blk, 0, stream>>>(xh, wqh, bq, nullptr, qh,
                                                  M_DIM, E_DIM, E_DIM, 0);
    gemm_f16_kernel<<<g_e, gemm_blk, 0, stream>>>(xh, wkh, bk, nullptr, kh,
                                                  M_DIM, E_DIM, E_DIM, 0);
    gemm_f16_kernel<<<g_e, gemm_blk, 0, stream>>>(xh, wvh, bv, nullptr, vh,
                                                  M_DIM, E_DIM, E_DIM, 0);

    // flash attention -> ctx (f16)
    attn_kernel<<<dim3(S_DIM / 64, H_DIM, B_DIM), dim3(128), 0, stream>>>(
        qh, kh, vh, pb, amask, ctxh);

    // output projection (fp32 out)
    gemm_f16_kernel<<<g_e, gemm_blk, 0, stream>>>(ctxh, woh, bo, attnf, nullptr,
                                                  M_DIM, E_DIM, E_DIM, 0);

    // h = LN(x + attn)
    add_ln_kernel<<<dim3(M_DIM), dim3(256), 0, stream>>>(x, attnf, ln1g, ln1b,
                                                         hf, hh);

    // FFN: gelu(h@W1+b1) @ W2 + b2
    gemm_f16_kernel<<<g_f, gemm_blk, 0, stream>>>(hh, w1h, b1, nullptr, f1h,
                                                  M_DIM, FF_DIM, E_DIM, 1);
    gemm_f16_kernel<<<g_e, gemm_blk, 0, stream>>>(f1h, w2h, b2, f2f, nullptr,
                                                  M_DIM, E_DIM, FF_DIM, 0);

    // out = LN(h + ffn)
    add_ln_kernel<<<dim3(M_DIM), dim3(256), 0, stream>>>(hf, f2f, ln2g, ln2b,
                                                         (float*)d_out, nullptr);
}
